// GNNGraph_28080496181821
// MI455X (gfx1250) — compile-verified, matrix operand order checked by reading, not run on hardware
//
#include <hip/hip_runtime.h>
#include <hip/hip_bf16.h>

// ---------------------------------------------------------------------------
// GAT (2x GATConv + pool + MLP head) for MI455X / gfx1250.
// Dense node-feature GEMMs use v_wmma_f32_16x16x32_bf16 (wave32 WMMA).
// Sparse edge phase uses coalesced fp32 atomics (L2-resident working set).
// ---------------------------------------------------------------------------

#define N_NODES  50000
#define N_EDGES  800000
#define E_TOT    (N_EDGES + N_NODES)   // self-loops appended
#define D_HID    64
#define N_CLASSES 10
#define N_GRAPHS  64
#define NEG_SLOPE 0.2f

typedef __attribute__((ext_vector_type(16))) __bf16 v16bf;
typedef __attribute__((ext_vector_type(8)))  float  v8f;

// ---------------------------------------------------------------------------
// C[M,64] = A[M,64] @ B[64,64]  (fp32 in/out, bf16 WMMA inner product)
// Block = 256 threads = 8 waves; each wave owns a 16-row strip (128 rows/blk).
// Per wave: 2 A-fragments (K=0..31, 32..63), 4 col tiles x 2 WMMAs each.
// Fragment layouts follow CDNA5 ISA (16-bit A 16x32, f32 C/D 16x16).
// Epilogue: wave-uniform tail check -> straight-line immediate-offset stores
// for all but the last partial block.
// ---------------------------------------------------------------------------
__global__ __launch_bounds__(256) void gat_gemm64_wmma(
    const float* __restrict__ A, const float* __restrict__ B,
    float* __restrict__ C, int M)
{
    const int lane = threadIdx.x & 31;
    const int wave = threadIdx.x >> 5;
    const int half = lane >> 4;      // 0: K sub-block low, 1: high
    const int lrow = lane & 15;      // row within tile (A) / col within tile (B)
    const int kb   = half * 8;
    const int rowBase = blockIdx.x * 128 + wave * 16;
    const bool fullTile = (rowBase + 16 <= M);   // wave-uniform

    int r = rowBase + lrow;
    if (r >= M) r = M - 1;           // clamp loads; stores are guarded
    const float* __restrict__ arow = A + (long)r * D_HID;

    // A fragments: lane half holds K = kb..kb+7 and 16+kb..16+kb+7 (per chunk)
    v16bf a0, a1;
#pragma unroll
    for (int j = 0; j < 8; ++j) {
        a0[j]     = (__bf16)arow[kb + j];
        a0[8 + j] = (__bf16)arow[16 + kb + j];
        a1[j]     = (__bf16)arow[32 + kb + j];
        a1[8 + j] = (__bf16)arow[48 + kb + j];
    }

#pragma unroll
    for (int t = 0; t < 4; ++t) {
        const int c = t * 16 + lrow;
        v16bf b0, b1;
#pragma unroll
        for (int j = 0; j < 8; ++j) {
            b0[j]     = (__bf16)B[(kb + j) * D_HID + c];
            b0[8 + j] = (__bf16)B[(16 + kb + j) * D_HID + c];
            b1[j]     = (__bf16)B[(32 + kb + j) * D_HID + c];
            b1[8 + j] = (__bf16)B[(48 + kb + j) * D_HID + c];
        }
        v8f acc = {};
        acc = __builtin_amdgcn_wmma_f32_16x16x32_bf16(
                  false, a0, false, b0, (short)0, acc, false, false);
        acc = __builtin_amdgcn_wmma_f32_16x16x32_bf16(
                  false, a1, false, b1, (short)0, acc, false, false);

        // C/D layout: VGPR i -> row = rowBase + 8*half + i, col = t*16 + lrow
        float* __restrict__ cp = C + (long)(rowBase + 8 * half) * D_HID + c;
        if (fullTile) {
#pragma unroll
            for (int i = 0; i < 8; ++i) cp[i * D_HID] = acc[i];
        } else {
#pragma unroll
            for (int i = 0; i < 8; ++i)
                if (rowBase + 8 * half + i < M) cp[i * D_HID] = acc[i];
        }
    }
}

// ---------------------------------------------------------------------------
// Per-node: alpha_src/dst dot products, init segment max/denom, zero agg row.
// ---------------------------------------------------------------------------
__global__ void gat_node_prep(const float* __restrict__ h,
                              const float* __restrict__ av_s,
                              const float* __restrict__ av_d,
                              float* __restrict__ as_, float* __restrict__ ad_,
                              float* __restrict__ m,  float* __restrict__ den,
                              float* __restrict__ agg, int n)
{
    int i = blockIdx.x * blockDim.x + threadIdx.x;
    if (i >= n) return;
    const float* hr = h + (long)i * D_HID;
    float s0 = 0.f, s1 = 0.f;
#pragma unroll 8
    for (int k = 0; k < D_HID; ++k) { float v = hr[k]; s0 += v * av_s[k]; s1 += v * av_d[k]; }
    as_[i] = s0;
    ad_[i] = s1;
    m[i]   = -__builtin_inff();
    den[i] = 0.f;
    float* ar = agg + (long)i * D_HID;
#pragma unroll 8
    for (int k = 0; k < D_HID; ++k) ar[k] = 0.f;
}

__device__ __forceinline__ void edge_sd(const int* __restrict__ ei, int e,
                                        int& s, int& d)
{
    if (e < N_EDGES) { s = ei[e]; d = ei[N_EDGES + e]; }
    else             { s = d = e - N_EDGES; }            // self loop
}

__device__ __forceinline__ float leaky(float x)
{
    return x > 0.f ? x : NEG_SLOPE * x;
}

// float atomic max via signed/unsigned bit ordering (init must be -inf)
__device__ __forceinline__ void atomicMaxFloat(float* addr, float v)
{
    if (v >= 0.f) atomicMax((int*)addr, __float_as_int(v));
    else          atomicMin((unsigned int*)addr, __float_as_uint(v));
}

__global__ void gat_edge_max(const int* __restrict__ ei,
                             const float* __restrict__ as_,
                             const float* __restrict__ ad_,
                             float* __restrict__ m)
{
    int e = blockIdx.x * blockDim.x + threadIdx.x;
    if (e >= E_TOT) return;
    int s, d; edge_sd(ei, e, s, d);
    atomicMaxFloat(&m[d], leaky(as_[s] + ad_[d]));
}

__global__ void gat_edge_exp(const int* __restrict__ ei,
                             const float* __restrict__ as_,
                             const float* __restrict__ ad_,
                             const float* __restrict__ m,
                             float* __restrict__ ex, float* __restrict__ den)
{
    int e = blockIdx.x * blockDim.x + threadIdx.x;
    if (e >= E_TOT) return;
    int s, d; edge_sd(ei, e, s, d);
    float v = __expf(leaky(as_[s] + ad_[d]) - m[d]);
    ex[e] = v;
    atomicAdd(&den[d], v);
}

// One edge handled by 32 consecutive threads, 2 dims each -> coalesced b64
// reads of h[src] and coalesced global_atomic_add_f32 pairs into agg[dst].
__global__ void gat_edge_scatter(const int* __restrict__ ei,
                                 const float* __restrict__ ex,
                                 const float* __restrict__ den,
                                 const float* __restrict__ h,
                                 float* __restrict__ agg)
{
    int gid = blockIdx.x * blockDim.x + threadIdx.x;
    if (gid >= E_TOT * (D_HID / 2)) return;
    int e = gid >> 5;
    int k = (gid & 31) * 2;
    int s, d; edge_sd(ei, e, s, d);
    float alpha = ex[e] / den[d];
    const float2 hv = *(const float2*)(h + (long)s * D_HID + k);
    float* ap = agg + (long)d * D_HID + k;
    atomicAdd(ap,     alpha * hv.x);
    atomicAdd(ap + 1, alpha * hv.y);
}

__global__ void gat_node_finish(float* __restrict__ agg,
                                const float* __restrict__ bias, int total)
{
    int i = blockIdx.x * blockDim.x + threadIdx.x;
    if (i >= total) return;
    float v = agg[i] + bias[i & 63];
    agg[i] = v > 0.f ? v : 0.f;
}

__global__ void gat_zero(float* __restrict__ p, int n)
{
    int i = blockIdx.x * blockDim.x + threadIdx.x;
    if (i < n) p[i] = 0.f;
}

__global__ void gat_pool(const float* __restrict__ h,
                         const int* __restrict__ batch,
                         float* __restrict__ g, int total)
{
    int i = blockIdx.x * blockDim.x + threadIdx.x;
    if (i >= total) return;
    int node = i >> 6;
    int k    = i & 63;
    atomicAdd(&g[(long)batch[node] * D_HID + k], h[i]);
}

// 64 graphs -> one thread each: fc1(64x64)+relu, fc2(64x10), log_softmax.
__global__ __launch_bounds__(64) void gat_head(const float* __restrict__ g,
                                               const float* __restrict__ fc1w,
                                               const float* __restrict__ fc1b,
                                               const float* __restrict__ fc2w,
                                               const float* __restrict__ fc2b,
                                               float* __restrict__ out)
{
    int t = threadIdx.x;
    if (t >= N_GRAPHS) return;
    const float* gr = g + (long)t * D_HID;

    float z[D_HID];
#pragma unroll 4
    for (int j = 0; j < D_HID; ++j) {
        float s = fc1b[j];
        for (int k = 0; k < D_HID; ++k) s += gr[k] * fc1w[k * D_HID + j];
        z[j] = s > 0.f ? s : 0.f;
    }

    float lo[N_CLASSES];
    float mx = -__builtin_inff();
#pragma unroll
    for (int c = 0; c < N_CLASSES; ++c) {
        float s = fc2b[c];
        for (int k = 0; k < D_HID; ++k) s += z[k] * fc2w[k * N_CLASSES + c];
        lo[c] = s;
        mx = s > mx ? s : mx;
    }
    float se = 0.f;
#pragma unroll
    for (int c = 0; c < N_CLASSES; ++c) se += __expf(lo[c] - mx);
    float lse = mx + __logf(se);
#pragma unroll
    for (int c = 0; c < N_CLASSES; ++c) out[t * N_CLASSES + c] = lo[c] - lse;
}

// ---------------------------------------------------------------------------
extern "C" void kernel_launch(void* const* d_in, const int* in_sizes, int n_in,
                              void* d_out, int out_size, void* d_ws, size_t ws_size,
                              hipStream_t stream)
{
    const float* x      = (const float*)d_in[0];
    const int*   ei     = (const int*)d_in[1];
    const int*   batch  = (const int*)d_in[2];
    const float* W1     = (const float*)d_in[3];
    const float* asrc1  = (const float*)d_in[4];
    const float* adst1  = (const float*)d_in[5];
    const float* b1     = (const float*)d_in[6];
    const float* W2     = (const float*)d_in[7];
    const float* asrc2  = (const float*)d_in[8];
    const float* adst2  = (const float*)d_in[9];
    const float* b2     = (const float*)d_in[10];
    const float* fc1w   = (const float*)d_in[11];
    const float* fc1b   = (const float*)d_in[12];
    const float* fc2w   = (const float*)d_in[13];
    const float* fc2b   = (const float*)d_in[14];
    float* out = (float*)d_out;

    // Workspace layout (floats); total ~7.45M floats ~= 29.8 MB
    const long NH = (long)N_NODES * D_HID;
    float* ws   = (float*)d_ws;
    float* h    = ws;                       // [N,64] post-GEMM features
    float* agg  = ws + NH;                  // [N,64] aggregation / next features
    float* as_  = ws + 2 * NH;              // [N]
    float* ad_  = as_ + N_NODES;            // [N]
    float* mbuf = ad_ + N_NODES;            // [N] segment max
    float* den  = mbuf + N_NODES;           // [N] segment sum
    float* ex   = den + N_NODES;            // [E_TOT]
    float* g    = ex + E_TOT;               // [64,64] pooled graph features

    const int gemmBlocks  = (N_NODES + 127) / 128;
    const int nodeBlocks  = (N_NODES + 255) / 256;
    const int elemBlocks  = (N_NODES * D_HID + 255) / 256;
    const int edgeBlocks  = (E_TOT + 255) / 256;
    const int scatBlocks  = (E_TOT * (D_HID / 2) + 255) / 256;

    // ---------------- layer 1 ----------------
    gat_gemm64_wmma<<<gemmBlocks, 256, 0, stream>>>(x, W1, h, N_NODES);
    gat_node_prep<<<nodeBlocks, 256, 0, stream>>>(h, asrc1, adst1, as_, ad_,
                                                  mbuf, den, agg, N_NODES);
    gat_edge_max<<<edgeBlocks, 256, 0, stream>>>(ei, as_, ad_, mbuf);
    gat_edge_exp<<<edgeBlocks, 256, 0, stream>>>(ei, as_, ad_, mbuf, ex, den);
    gat_edge_scatter<<<scatBlocks, 256, 0, stream>>>(ei, ex, den, h, agg);
    gat_node_finish<<<elemBlocks, 256, 0, stream>>>(agg, b1, N_NODES * D_HID);

    // ---------------- layer 2 (agg -> h -> agg) ----------------
    gat_gemm64_wmma<<<gemmBlocks, 256, 0, stream>>>(agg, W2, h, N_NODES);
    gat_node_prep<<<nodeBlocks, 256, 0, stream>>>(h, asrc2, adst2, as_, ad_,
                                                  mbuf, den, agg, N_NODES);
    gat_edge_max<<<edgeBlocks, 256, 0, stream>>>(ei, as_, ad_, mbuf);
    gat_edge_exp<<<edgeBlocks, 256, 0, stream>>>(ei, as_, ad_, mbuf, ex, den);
    gat_edge_scatter<<<scatBlocks, 256, 0, stream>>>(ei, ex, den, h, agg);
    gat_node_finish<<<elemBlocks, 256, 0, stream>>>(agg, b2, N_NODES * D_HID);

    // ---------------- pool + head ----------------
    gat_zero<<<(N_GRAPHS * D_HID + 255) / 256, 256, 0, stream>>>(g, N_GRAPHS * D_HID);
    gat_pool<<<elemBlocks, 256, 0, stream>>>(agg, batch, g, N_NODES * D_HID);
    gat_head<<<1, 64, 0, stream>>>(g, fc1w, fc1b, fc2w, fc2b, out);
}
